// Wrapped8BitLinear_57466662420641
// MI455X (gfx1250) — 8-bit matrix operand order repaired and checked by reading, compiled, not run on hardware
//
#include <hip/hip_runtime.h>
#include <hip/hip_fp16.h>
#include <stdint.h>

typedef int v8i __attribute__((ext_vector_type(8)));

#define THRESHOLD 6.0f
#define KO_MAX 128

__device__ __forceinline__ void async_ld_b128(uint32_t lds_addr, const void* gaddr) {
    asm volatile("global_load_async_to_lds_b128 %0, %1, off" ::"v"(lds_addr), "v"(gaddr)
                 : "memory");
}
__device__ __forceinline__ void wait_async0() {
    asm volatile("s_wait_asynccnt 0x0" ::: "memory");
}

__global__ void init_mask_kernel(uint32_t* mask, int maskWords) {
    int i = blockIdx.x * blockDim.x + threadIdx.x;
    if (i < maskWords) mask[i] = 0u;
}

__global__ void colmask_kernel(const __half* __restrict__ x, uint32_t* __restrict__ mask,
                               int N, int DIN) {
    int c = blockIdx.x * 256 + threadIdx.x;
    int r0 = blockIdx.y * 64;
    if (c >= DIN) return;
    bool flag = false;
    for (int r = 0; r < 64; ++r) {
        int row = r0 + r;
        if (row < N) {
            float v = __half2float(x[(size_t)row * DIN + c]);
            flag |= (fabsf(v) > THRESHOLD);
        }
    }
    if (flag) atomicOr(&mask[c >> 5], 1u << (c & 31));
}

__global__ void build_list_kernel(const uint32_t* __restrict__ mask, int* __restrict__ list,
                                  int* __restrict__ cnt, int maskWords) {
    if (threadIdx.x != 0 || blockIdx.x != 0) return;
    int n = 0;
    for (int w = 0; w < maskWords; ++w) {
        uint32_t b = mask[w];
        while (b) {
            int t = __ffs(b) - 1;
            b &= b - 1;
            if (n < KO_MAX) list[n] = w * 32 + t;
            ++n;
        }
    }
    *cnt = (n > KO_MAX) ? KO_MAX : n;
}

__global__ void quantX_kernel(const __half* __restrict__ x, const uint32_t* __restrict__ mask,
                              int8_t* __restrict__ Xq, float* __restrict__ sx, int DIN) {
    int row = blockIdx.x;
    int tx = threadIdx.x;
    const __half* xr = x + (size_t)row * DIN;
    __shared__ float red[256];
    float m = 0.f;
    for (int c = tx; c < DIN; c += 256) {
        uint32_t bit = (mask[c >> 5] >> (c & 31)) & 1u;
        float v = __half2float(xr[c]);
        if (!bit) m = fmaxf(m, fabsf(v));
    }
    red[tx] = m;
    __syncthreads();
    for (int s = 128; s > 0; s >>= 1) {
        if (tx < s) red[tx] = fmaxf(red[tx], red[tx + s]);
        __syncthreads();
    }
    float mx = fmaxf(red[0], 1e-8f);
    if (tx == 0) sx[row] = mx;
    float inv = 127.f / mx;
    for (int c = tx; c < DIN; c += 256) {
        uint32_t bit = (mask[c >> 5] >> (c & 31)) & 1u;
        float v = __half2float(xr[c]);
        float q = bit ? 0.f : fminf(fmaxf(rintf(v * inv), -127.f), 127.f);
        Xq[(size_t)row * DIN + c] = (int8_t)q;
    }
}

__global__ void quantW_kernel(const __half* __restrict__ CB, const __half* __restrict__ SCB,
                              int8_t* __restrict__ Wq, float* __restrict__ sw, int DIN) {
    int row = blockIdx.x;
    int tx = threadIdx.x;
    const __half* cbr = CB + (size_t)row * DIN;
    float scb = __half2float(SCB[row]) * (1.f / 127.f);
    __shared__ float red[256];
    float m = 0.f;
    for (int c = tx; c < DIN; c += 256) {
        float w = __half2float(cbr[c]) * scb;
        m = fmaxf(m, fabsf(w));
    }
    red[tx] = m;
    __syncthreads();
    for (int s = 128; s > 0; s >>= 1) {
        if (tx < s) red[tx] = fmaxf(red[tx], red[tx + s]);
        __syncthreads();
    }
    float mx = fmaxf(red[0], 1e-8f);
    if (tx == 0) sw[row] = mx;
    float inv = 127.f / mx;
    for (int c = tx; c < DIN; c += 256) {
        float w = __half2float(cbr[c]) * scb;
        float q = fminf(fmaxf(rintf(w * inv), -127.f), 127.f);
        Wq[(size_t)row * DIN + c] = (int8_t)q;
    }
}

__global__ void gatherWc_kernel(const __half* __restrict__ CB, const __half* __restrict__ SCB,
                                const int* __restrict__ list, const int* __restrict__ cnt,
                                float* __restrict__ Wc, int DOUT, int DIN) {
    int o = blockIdx.y;
    if (o >= *cnt) return;
    int m = blockIdx.x * 256 + threadIdx.x;
    if (m >= DOUT) return;
    int c = list[o];
    Wc[(size_t)o * DOUT + m] =
        __half2float(CB[(size_t)m * DIN + c]) * __half2float(SCB[m]) * (1.f / 127.f);
}

__global__ __launch_bounds__(256) void gemm_int8_kernel(
    const int8_t* __restrict__ Xq, const int8_t* __restrict__ Wq,
    const float* __restrict__ sx, const float* __restrict__ sw,
    const float* __restrict__ Wc, const int* __restrict__ list, const int* __restrict__ cnt,
    const __half* __restrict__ x, const __half* __restrict__ bias,
    __half* __restrict__ out, int DOUT, int DIN) {

    constexpr int BK = 64;
    constexpr int LDT = 80;
    constexpr int BUF = 128 * LDT;
    __shared__ __align__(16) uint8_t As[2][BUF];
    __shared__ __align__(16) uint8_t Bs[2][BUF];

    const int tid = threadIdx.x;
    const int lane = tid & 31;
    const int wave = tid >> 5;
    const int wm = wave & 1;
    const int wn = wave >> 1;
    const int half = lane >> 4;
    const int l16 = lane & 15;
    const int m0 = blockIdx.y * 128;
    const int n0 = blockIdx.x * 128;

    const uint32_t asBase = (uint32_t)(uintptr_t)&As[0][0];
    const uint32_t bsBase = (uint32_t)(uintptr_t)&Bs[0][0];

    const int rA0 = tid >> 2, sA0 = (tid & 3) * 16;
    const int rA1 = (tid + 256) >> 2, sA1 = ((tid + 256) & 3) * 16;

    const v8i zero = {0, 0, 0, 0, 0, 0, 0, 0};
    v8i acc[4][2];
#pragma unroll
    for (int i = 0; i < 4; ++i)
#pragma unroll
        for (int j = 0; j < 2; ++j) acc[i][j] = zero;

    {
        const int k0 = 0;
        async_ld_b128(asBase + rA0 * LDT + sA0, Xq + (size_t)(m0 + rA0) * DIN + k0 + sA0);
        async_ld_b128(asBase + rA1 * LDT + sA1, Xq + (size_t)(m0 + rA1) * DIN + k0 + sA1);
        async_ld_b128(bsBase + rA0 * LDT + sA0, Wq + (size_t)(n0 + rA0) * DIN + k0 + sA0);
        async_ld_b128(bsBase + rA1 * LDT + sA1, Wq + (size_t)(n0 + rA1) * DIN + k0 + sA1);
    }

    const int nk = DIN / BK;
    for (int kt = 0; kt < nk; ++kt) {
        const int cur = kt & 1;
        wait_async0();
        __syncthreads();

        if (kt + 1 < nk) {
            const int k0 = (kt + 1) * BK;
            const uint32_t ao = asBase + (cur ^ 1) * BUF;
            const uint32_t bo = bsBase + (cur ^ 1) * BUF;
            async_ld_b128(ao + rA0 * LDT + sA0, Xq + (size_t)(m0 + rA0) * DIN + k0 + sA0);
            async_ld_b128(ao + rA1 * LDT + sA1, Xq + (size_t)(m0 + rA1) * DIN + k0 + sA1);
            async_ld_b128(bo + rA0 * LDT + sA0, Wq + (size_t)(n0 + rA0) * DIN + k0 + sA0);
            async_ld_b128(bo + rA1 * LDT + sA1, Wq + (size_t)(n0 + rA1) * DIN + k0 + sA1);
        }

        const uint8_t* Ac = As[cur];
        const uint8_t* Bc = Bs[cur];

        v8i af[4];
#pragma unroll
        for (int mi = 0; mi < 4; ++mi) {
            const uint2* p = (const uint2*)(Ac + (wm * 64 + mi * 16 + l16) * LDT + half * 16);
            v8i a;
            uint2 t;
            t = p[0]; a[0] = (int)t.x; a[1] = (int)t.y;
            t = p[1]; a[2] = (int)t.x; a[3] = (int)t.y;
            t = p[4]; a[4] = (int)t.x; a[5] = (int)t.y;
            t = p[5]; a[6] = (int)t.x; a[7] = (int)t.y;
            af[mi] = a;
        }
        v8i bf[2];
#pragma unroll
        for (int ni = 0; ni < 2; ++ni) {
            const uint8_t* base = Bc + (wn * 32 + ni * 16 + l16) * LDT + half * 16;
            uint4 q0 = *(const uint4*)(base);
            uint4 q1 = *(const uint4*)(base + 32);
            v8i b;
            b[0] = (int)q0.x; b[1] = (int)q0.y; b[2] = (int)q0.z; b[3] = (int)q0.w;
            b[4] = (int)q1.x; b[5] = (int)q1.y; b[6] = (int)q1.z; b[7] = (int)q1.w;
            bf[ni] = b;
        }
#pragma unroll
        for (int mi = 0; mi < 4; ++mi)
#pragma unroll
            for (int ni = 0; ni < 2; ++ni)
                acc[mi][ni] = __builtin_amdgcn_wmma_i32_16x16x64_iu8(
                    true, af[mi], true, bf[ni], acc[mi][ni], false, false);
    }

    const int ko = *cnt;
#pragma unroll
    for (int mi = 0; mi < 4; ++mi) {
#pragma unroll
        for (int ni = 0; ni < 2; ++ni) {
            int dout = n0 + wn * 32 + ni * 16 + l16;
            float swv = sw[dout];
            float bv = __half2float(bias[dout]);
#pragma unroll
            for (int v = 0; v < 8; ++v) {
                int token = m0 + wm * 64 + mi * 16 + v + 8 * half;
                float f = (float)acc[mi][ni][v] * sx[token] * swv * (1.f / 16129.f);
                for (int o = 0; o < ko; ++o) {
                    int c = list[o];
                    f += __half2float(x[(size_t)token * DIN + c]) * Wc[(size_t)o * DOUT + dout];
                }
                f += bv;
                out[(size_t)token * DOUT + dout] = __float2half(f);
            }
        }
    }
}

extern "C" void kernel_launch(void* const* d_in, const int* in_sizes, int n_in,
                              void* d_out, int out_size, void* d_ws, size_t ws_size,
                              hipStream_t stream) {
    (void)n_in; (void)out_size; (void)ws_size;
    const __half* x    = (const __half*)d_in[0];
    const __half* CB   = (const __half*)d_in[1];
    const __half* SCB  = (const __half*)d_in[2];
    const __half* bias = (const __half*)d_in[3];
    __half* out = (__half*)d_out;

    const int DOUT = in_sizes[3];
    const int DIN  = in_sizes[1] / DOUT;
    const int N    = in_sizes[0] / DIN;
    const int maskWords = DIN / 32;

    uint8_t* ws = (uint8_t*)d_ws;
    size_t off = 0;
    auto carve = [&](size_t bytes) -> uint8_t* {
        uint8_t* p = ws + off;
        off = (off + bytes + 255) & ~(size_t)255;
        return p;
    };
    int8_t*   Xq   = (int8_t*)carve((size_t)N * DIN);
    int8_t*   Wq   = (int8_t*)carve((size_t)DOUT * DIN);
    float*    sx   = (float*)carve((size_t)N * 4);
    float*    sw   = (float*)carve((size_t)DOUT * 4);
    uint32_t* mask = (uint32_t*)carve((size_t)maskWords * 4);
    int*      cnt  = (int*)carve(256);
    int*      list = (int*)carve((size_t)KO_MAX * 4);
    float*    Wc   = (float*)carve((size_t)KO_MAX * DOUT * 4);

    init_mask_kernel<<<(maskWords + 255) / 256, 256, 0, stream>>>(mask, maskWords);
    colmask_kernel<<<dim3((DIN + 255) / 256, (N + 63) / 64), 256, 0, stream>>>(x, mask, N, DIN);
    build_list_kernel<<<1, 1, 0, stream>>>(mask, list, cnt, maskWords);
    quantX_kernel<<<N, 256, 0, stream>>>(x, mask, Xq, sx, DIN);
    quantW_kernel<<<DOUT, 256, 0, stream>>>(CB, SCB, Wq, sw, DIN);
    gatherWc_kernel<<<dim3((DOUT + 255) / 256, KO_MAX), 256, 0, stream>>>(CB, SCB, list, cnt, Wc,
                                                                          DOUT, DIN);
    gemm_int8_kernel<<<dim3(DOUT / 128, N / 128), 256, 0, stream>>>(
        Xq, Wq, sx, sw, Wc, list, cnt, x, bias, out, DOUT, DIN);
}
